// RelativeMultiHeadAttention_4904852652828
// MI455X (gfx1250) — matrix operand order repaired and checked by reading, not compiled, not run on hardware
//
#include <hip/hip_runtime.h>
#include <hip/hip_bf16.h>
#include <math.h>

// ---- problem constants ----
#define B_    16
#define N_    1024
#define D_    256
#define H_    8
#define DK_   32
#define NREL_ 3969   // (2*32-1)^2

typedef __attribute__((ext_vector_type(16))) __bf16 v16bf;
typedef __attribute__((ext_vector_type(8)))  __bf16 v8bf;
typedef __attribute__((ext_vector_type(8)))  float  v8f;

__device__ __forceinline__ v8f wmma_bf16(v16bf a, v16bf b, v8f c) {
    // D(16x16,f32) = A(16x32,bf16) * B(32x16,bf16) + C
    return __builtin_amdgcn_wmma_f32_16x16x32_bf16(
        /*neg_a=*/false, a, /*neg_b=*/false, b,
        /*c_mod=*/(short)0, c, /*reuse_a=*/false, /*reuse_b=*/false);
}

__device__ __forceinline__ v16bf merge8(v8bf lo, v8bf hi) {
    v16bf r;
#pragma unroll
    for (int i = 0; i < 8; ++i) { r[i] = lo[i]; r[i + 8] = hi[i]; }
    return r;
}

// reductions across the 16 lanes of one half-wave (masks < 16 stay in group)
__device__ __forceinline__ float hmax16(float v) {
#pragma unroll
    for (int m = 1; m < 16; m <<= 1) v = fmaxf(v, __shfl_xor(v, m, 32));
    return v;
}
__device__ __forceinline__ float hsum16(float v) {
#pragma unroll
    for (int m = 1; m < 16; m <<= 1) v += __shfl_xor(v, m, 32);
    return v;
}

// ---------------- fp32 -> bf16 conversion ----------------
__global__ void cvt_bf16_kernel(const float* __restrict__ src,
                                __bf16* __restrict__ dst, int n) {
    int i = blockIdx.x * blockDim.x + threadIdx.x;
    int stride = gridDim.x * blockDim.x;
    for (; i < n; i += stride) dst[i] = (__bf16)src[i];
}

// ---------------- rel bias gather: [H,N,N] f32 (batch-shared) ----------------
__global__ void relbias_kernel(const float* __restrict__ table,
                               const int* __restrict__ relidx,
                               float* __restrict__ out) {
    int i = blockIdx.x * blockDim.x + threadIdx.x;
    int stride = gridDim.x * blockDim.x;
    const int total = H_ * N_ * N_;
    for (; i < total; i += stride) {
        int h = i / (N_ * N_);
        int pos = i - h * (N_ * N_);
        out[i] = table[h * NREL_ + relidx[pos]];
    }
}

// ---------------- fused QKV projection (bf16 WMMA, f32 acc) ----------------
// one wave -> one (16-token x 32-channel) tile of one projection.
// wid = p*8192 + tt*8 + cc   (p in {q,k,v}, tt token tile 0..1023, cc = head)
__global__ void qkv_kernel(const __bf16* __restrict__ xb,
                           const __bf16* __restrict__ wq,
                           const __bf16* __restrict__ wk,
                           const __bf16* __restrict__ wv,
                           const float* __restrict__ bq,
                           const float* __restrict__ bk,
                           const float* __restrict__ bv,
                           __bf16* __restrict__ Qb,   // [b,h,tok,dk]
                           __bf16* __restrict__ Kb,   // [b,h,key,dk]
                           __bf16* __restrict__ Vtb)  // [b,h,dk,tok]
{
    const int lane = threadIdx.x & 31;
    const int wid  = blockIdx.x * (blockDim.x >> 5) + (threadIdx.x >> 5);
    const int p    = wid >> 13;          // 0=Q 1=K 2=V
    const int rem  = wid & 8191;
    const int tt   = rem >> 3;           // token tile (16 tokens)
    const int cc   = rem & 7;            // head index (32 channels)
    const int half = lane >> 4;
    const int nlo  = lane & 15;
    const int c8   = 8 * half;

    const __bf16* W    = (p == 0) ? wq : (p == 1) ? wk : wv;
    const float*  bias = (p == 0) ? bq : (p == 1) ? bk : bv;

    const int arow = tt * 16 + nlo;      // A row = token
    const int chlo = cc * 32 + nlo;      // output channel (lo acc)
    const int chhi = chlo + 16;

    v8f acc_lo = {}; v8f acc_hi = {};
#pragma unroll 2
    for (int kk = 0; kk < D_; kk += 32) {
        v8bf a0 = *(const v8bf*)(xb + arow * D_ + kk + 16 * half);
        v8bf a1 = *(const v8bf*)(xb + arow * D_ + kk + 16 * half + 8);
        v16bf a = merge8(a0, a1);
        v16bf blo = *(const v16bf*)(W + chlo * D_ + kk + 16 * half);
        v16bf bhi = *(const v16bf*)(W + chhi * D_ + kk + 16 * half);
        acc_lo = wmma_bf16(a, blo, acc_lo);
        acc_hi = wmma_bf16(a, bhi, acc_hi);
    }
    const float bs_lo = bias[chlo];
    const float bs_hi = bias[chhi];

#pragma unroll
    for (int r = 0; r < 8; ++r) {
        int tg  = tt * 16 + r + 8 * half;     // global token
        int b   = tg >> 10;
        int tok = tg & (N_ - 1);
        float vlo = acc_lo[r] + bs_lo;
        float vhi = acc_hi[r] + bs_hi;
        if (p == 0) {
            __bf16* qp = Qb + ((size_t)(b * H_ + cc) * N_ + tok) * DK_;
            qp[nlo]      = (__bf16)vlo;
            qp[16 + nlo] = (__bf16)vhi;
        } else if (p == 1) {
            __bf16* kp = Kb + ((size_t)(b * H_ + cc) * N_ + tok) * DK_;
            kp[nlo]      = (__bf16)vlo;
            kp[16 + nlo] = (__bf16)vhi;
        } else {
            __bf16* vp = Vtb + (size_t)(b * H_ + cc) * DK_ * N_;
            vp[(size_t)nlo * N_ + tok]        = (__bf16)vlo;  // contiguous in tok over r
            vp[(size_t)(16 + nlo) * N_ + tok] = (__bf16)vhi;
        }
    }
}

// ---------------- flash attention: one wave per (b,h,16-query tile) --------
__global__ void attn_kernel(const __bf16* __restrict__ Qb,
                            const __bf16* __restrict__ Kb,
                            const __bf16* __restrict__ Vtb,
                            const float* __restrict__ relb,  // [H,N,N]
                            __bf16* __restrict__ Ob)         // [b,tok,D]
{
    __shared__ __bf16 plds[8][16 * 40];   // per-wave 16x32 P tile, row stride 40 (16B aligned)
    const int lane  = threadIdx.x & 31;
    const int wslot = threadIdx.x >> 5;
    const int wid   = blockIdx.x * 8 + wslot;
    const int qt = wid & 63;
    const int bh = wid >> 6;
    const int b  = bh >> 3;
    const int h  = bh & 7;
    const int half = lane >> 4;
    const int nlo  = lane & 15;
    const int c8   = 8 * half;
    const int q0   = qt * 16;

    const __bf16* Qbase = Qb  + (size_t)bh * N_ * DK_;
    const __bf16* Kbase = Kb  + (size_t)bh * N_ * DK_;
    const __bf16* Vbase = Vtb + (size_t)bh * DK_ * N_;
    const float*  rb    = relb + (size_t)h * N_ * N_;

    // A operand: 16x32 query tile (resident for whole loop)
    v8bf qa0 = *(const v8bf*)(Qbase + (q0 + nlo) * DK_ + 16 * half);
    v8bf qa1 = *(const v8bf*)(Qbase + (q0 + nlo) * DK_ + 16 * half + 8);
    v16bf qa = merge8(qa0, qa1);

    float run_max[8], run_sum[8];
#pragma unroll
    for (int r = 0; r < 8; ++r) { run_max[r] = -INFINITY; run_sum[r] = 0.f; }
    v8f o_lo = {}, o_hi = {};

    const float scale = 0.17677669529663687f;  // 1/sqrt(32)
    __bf16* P = &plds[wslot][0];
    const v8f zc = {};

    for (int k0 = 0; k0 < N_; k0 += 32) {
        if (k0 + 32 < N_) {  // uniform-branch prefetch of next K tile
            __builtin_prefetch(Kbase + (k0 + 32 + nlo) * DK_, 0, 1);
        }
        // scores: S = Q(16x32) * K^T(32x16) for two 16-key tiles
        v16bf bk0 = *(const v16bf*)(Kbase + (k0 + nlo) * DK_ + 16 * half);
        v16bf bk1 = *(const v16bf*)(Kbase + (k0 + 16 + nlo) * DK_ + 16 * half);
        v8f s0 = wmma_bf16(qa, bk0, zc);
        v8f s1 = wmma_bf16(qa, bk1, zc);

        // scale + relative position bias
#pragma unroll
        for (int r = 0; r < 8; ++r) {
            int qrow = q0 + r + 8 * half;
            float b0 = rb[(size_t)qrow * N_ + k0 + nlo];
            float b1 = rb[(size_t)qrow * N_ + k0 + 16 + nlo];
            s0[r] = s0[r] * scale + b0;
            s1[r] = s1[r] * scale + b1;
        }

        // online softmax (per query row; row lives in one 16-lane half)
#pragma unroll
        for (int r = 0; r < 8; ++r) {
            float mx = hmax16(fmaxf(s0[r], s1[r]));
            float nm = fmaxf(run_max[r], mx);
            float corr = __expf(run_max[r] - nm);
            float p0 = __expf(s0[r] - nm);
            float p1 = __expf(s1[r] - nm);
            float rs = hsum16(p0 + p1);
            run_sum[r] = run_sum[r] * corr + rs;
            run_max[r] = nm;
            o_lo[r] *= corr; o_hi[r] *= corr;
            s0[r] = p0; s1[r] = p1;
        }

        // transpose P (acc layout -> A-operand layout) via per-wave LDS bounce
#pragma unroll
        for (int r = 0; r < 8; ++r) {
            int row = r + 8 * half;
            P[row * 40 + nlo]      = (__bf16)s0[r];
            P[row * 40 + 16 + nlo] = (__bf16)s1[r];
        }
        v8bf pa0 = *(const v8bf*)(P + nlo * 40 + 16 * half);       // DS in-order per wave
        v8bf pa1 = *(const v8bf*)(P + nlo * 40 + 16 * half + 8);
        v16bf pa = merge8(pa0, pa1);

        // O += P(16x32) * V(32x32): V^T layout makes B loads contiguous
        v16bf bv_lo = *(const v16bf*)(Vbase + (size_t)nlo * N_ + k0 + 16 * half);
        v16bf bv_hi = *(const v16bf*)(Vbase + (size_t)(16 + nlo) * N_ + k0 + 16 * half);
        o_lo = wmma_bf16(pa, bv_lo, o_lo);
        o_hi = wmma_bf16(pa, bv_hi, o_hi);
    }

    // normalize and store [b, tok, h*32 + dk] as bf16 for the output GEMM
#pragma unroll
    for (int r = 0; r < 8; ++r) {
        float inv = 1.0f / run_sum[r];
        int tok = q0 + r + 8 * half;
        __bf16* op = Ob + ((size_t)b * N_ + tok) * D_ + h * DK_;
        op[nlo]      = (__bf16)(o_lo[r] * inv);
        op[16 + nlo] = (__bf16)(o_hi[r] * inv);
    }
}

// ---------------- output projection: out = attn @ Wo^T + bo (f32 out) ------
__global__ void oproj_kernel(const __bf16* __restrict__ ab,
                             const __bf16* __restrict__ wo,
                             const float* __restrict__ bo,
                             float* __restrict__ out) {
    const int lane = threadIdx.x & 31;
    const int wid  = blockIdx.x * (blockDim.x >> 5) + (threadIdx.x >> 5);
    const int tt   = wid >> 3;           // token tile
    const int cc   = wid & 7;            // 32-channel chunk
    const int half = lane >> 4;
    const int nlo  = lane & 15;
    const int c8   = 8 * half;

    const int arow = tt * 16 + nlo;
    const int chlo = cc * 32 + nlo;
    const int chhi = chlo + 16;

    v8f acc_lo = {}; v8f acc_hi = {};
#pragma unroll 2
    for (int kk = 0; kk < D_; kk += 32) {
        v8bf a0 = *(const v8bf*)(ab + arow * D_ + kk + 16 * half);
        v8bf a1 = *(const v8bf*)(ab + arow * D_ + kk + 16 * half + 8);
        v16bf a = merge8(a0, a1);
        v16bf blo = *(const v16bf*)(wo + chlo * D_ + kk + 16 * half);
        v16bf bhi = *(const v16bf*)(wo + chhi * D_ + kk + 16 * half);
        acc_lo = wmma_bf16(a, blo, acc_lo);
        acc_hi = wmma_bf16(a, bhi, acc_hi);
    }
    const float bs_lo = bo[chlo];
    const float bs_hi = bo[chhi];
#pragma unroll
    for (int r = 0; r < 8; ++r) {
        size_t tg = (size_t)(tt * 16 + r + 8 * half);
        out[tg * D_ + chlo] = acc_lo[r] + bs_lo;
        out[tg * D_ + chhi] = acc_hi[r] + bs_hi;
    }
}

// ---------------- host side ----------------
extern "C" void kernel_launch(void* const* d_in, const int* in_sizes, int n_in,
                              void* d_out, int out_size, void* d_ws, size_t ws_size,
                              hipStream_t stream) {
    const float* x     = (const float*)d_in[0];
    const float* Wq    = (const float*)d_in[1];
    const float* bq    = (const float*)d_in[2];
    const float* Wk    = (const float*)d_in[3];
    const float* bk    = (const float*)d_in[4];
    const float* Wv    = (const float*)d_in[5];
    const float* bv    = (const float*)d_in[6];
    const float* Wo    = (const float*)d_in[7];
    const float* bo    = (const float*)d_in[8];
    const float* table = (const float*)d_in[9];
    const int*   ridx  = (const int*)d_in[10];
    float* out = (float*)d_out;

    char* ws = (char*)d_ws;
    size_t off = 0;
    auto alloc = [&](size_t bytes) -> void* {
        void* p = ws + off;
        off += (bytes + 255) & ~(size_t)255;
        return p;
    };
    __bf16* xb   = (__bf16*)alloc((size_t)B_ * N_ * D_ * 2);          // 8 MB
    __bf16* wqb  = (__bf16*)alloc((size_t)D_ * D_ * 2);
    __bf16* wkb  = (__bf16*)alloc((size_t)D_ * D_ * 2);
    __bf16* wvb  = (__bf16*)alloc((size_t)D_ * D_ * 2);
    __bf16* wob  = (__bf16*)alloc((size_t)D_ * D_ * 2);
    __bf16* Qb   = (__bf16*)alloc((size_t)B_ * H_ * N_ * DK_ * 2);    // 8 MB
    __bf16* Kb   = (__bf16*)alloc((size_t)B_ * H_ * N_ * DK_ * 2);    // 8 MB
    __bf16* Vtb  = (__bf16*)alloc((size_t)B_ * H_ * N_ * DK_ * 2);    // 8 MB
    __bf16* attb = (__bf16*)alloc((size_t)B_ * N_ * D_ * 2);          // 8 MB
    float*  relb = (float*) alloc((size_t)H_ * N_ * N_ * 4);          // 32 MB

    cvt_bf16_kernel<<<1024, 256, 0, stream>>>(x,  xb,  B_ * N_ * D_);
    cvt_bf16_kernel<<<64,   256, 0, stream>>>(Wq, wqb, D_ * D_);
    cvt_bf16_kernel<<<64,   256, 0, stream>>>(Wk, wkb, D_ * D_);
    cvt_bf16_kernel<<<64,   256, 0, stream>>>(Wv, wvb, D_ * D_);
    cvt_bf16_kernel<<<64,   256, 0, stream>>>(Wo, wob, D_ * D_);
    relbias_kernel<<<4096,  256, 0, stream>>>(table, ridx, relb);

    // 3 projections * 1024 token tiles * 8 heads = 24576 waves / 8 per block
    qkv_kernel<<<3072, 256, 0, stream>>>(xb, wqb, wkb, wvb, bq, bk, bv, Qb, Kb, Vtb);
    // 16*8*64 = 8192 waves / 8 per block
    attn_kernel<<<1024, 256, 0, stream>>>(Qb, Kb, Vtb, relb, attb);
    // 1024*8 = 8192 waves / 8 per block
    oproj_kernel<<<1024, 256, 0, stream>>>(attb, wob, bo, out);
}
